// TwoHemiRNNTanh_asymmetric_single_readout_51960514347822
// MI455X (gfx1250) — matrix operand order repaired and checked by reading, not compiled, not run on hardware
//
#include <hip/hip_runtime.h>
#include <hip/hip_bf16.h>
#include <math.h>
#include <stdint.h>

// ---------------------------------------------------------------------------
// TwoHemiRNNTanh on MI455X (gfx1250):
//  * W_full (258x258) lives in LDS as bf16 (transposed for B-fragments)
//  * 32 workgroups x 16 batch rows; each runs all 400 timesteps locally
//  * per-step GEMM via v_wmma_f32_16x16x32_bf16, f32 accumulate
//  * hidden state double-buffered in LDS -> fused epilogue, no acc spills,
//    single barrier per timestep
//  * deterministic hash-based Gaussian noise (stand-in for jax threefry)
// ---------------------------------------------------------------------------

typedef __attribute__((ext_vector_type(16))) __bf16 v16bf;
typedef __attribute__((ext_vector_type(8)))  __bf16 v8bf;
typedef __attribute__((ext_vector_type(8)))  float  v8f;

#define B_TOT   512
#define T_TOT   400
#define N_R     256
#define H_DIM   258          // N + 2
#define KP      288          // padded hidden dim (K and N of the GEMM)
#define B_TILE  16           // batch rows per workgroup (= WMMA M)
#define NT      17           // useful 16-wide N tiles (n0 = 0..256)
#define NWAVES  8            // 256 threads / wave32
#define A_LEAK  0.2f
#define SCL_NOISE 0.15811388300841897f   // sqrt(2/A) * 0.05
#define HSLOT   (B_TILE * KP)            // elements per h buffer

// LDS layout (bytes)
#define OFF_WT    0                      // KP*KP bf16            = 165888
#define OFF_HBF   165888                 // 2 * B_TILE*KP bf16    = 18432
#define OFF_HF    184320                 // 2 * B_TILE*KP f32     = 36864
#define OFF_BIAS  221184                 // KP f32                = 1152
#define OFF_WRO   222336                 // KP f32                = 1152
#define OFF_WXR   223488                 // 256*2 f32             = 2048
#define OFF_WXL   225536                 // 4 f32                 = 16
#define OFF_XL    225552                 // 2 * 32 f32            = 256
#define OFF_XR    225808                 // 2 * 32 f32            = 256
#define SMEM_BYTES 226064

__device__ __forceinline__ uint32_t hash_u32(uint32_t x) {
    x ^= x >> 16; x *= 0x7feb352dU;
    x ^= x >> 15; x *= 0x846ca68bU;
    x ^= x >> 16;
    return x;
}

__device__ __forceinline__ float u01(uint32_t h) {
    return (float)(h >> 8) * (1.0f / 16777216.0f) + (0.5f / 16777216.0f);
}

// deterministic standard normal from a 32-bit counter key (Box-Muller)
__device__ __forceinline__ float gauss(uint32_t key) {
    uint32_t h1 = hash_u32(key * 2654435761u + 0x9e3779b9u);
    uint32_t h2 = hash_u32(h1 ^ 0x85ebca6bu);
    float r = __fsqrt_rn(-2.0f * __logf(u01(h1)));
    return r * __cosf(6.28318530717958647f * u01(h2));
}

// branchless tanh: 1 - 2/(exp(2x)+1); exact at +-inf, no libcall
__device__ __forceinline__ float fast_tanh(float x) {
    float e = __expf(2.0f * x);
    return 1.0f - 2.0f / (e + 1.0f);
}

__global__ void __launch_bounds__(256, 1)
two_hemi_rnn_kernel(const float* __restrict__ xs,
                    const float* __restrict__ w_ll,
                    const float* __restrict__ b_ll,
                    const float* __restrict__ w_rr,
                    const float* __restrict__ b_rr,
                    const float* __restrict__ w_lr,
                    const float* __restrict__ w_rl,
                    const float* __restrict__ w_xl,
                    const float* __restrict__ w_xr,
                    const float* __restrict__ w_ro,
                    const float* __restrict__ b_ro,
                    float* __restrict__ out) {
    extern __shared__ char smem[];
    __bf16* WT    = (__bf16*)(smem + OFF_WT);   // WT[k*KP + j] = W_full[j][k]
    __bf16* h_bf  = (__bf16*)(smem + OFF_HBF);  // 2 buffers, bf16 A-matrix feed
    float*  h_f   = (float*)(smem + OFF_HF);    // 2 buffers, f32 master copy
    float*  bias  = (float*)(smem + OFF_BIAS);
    float*  wro   = (float*)(smem + OFF_WRO);
    float*  wxr   = (float*)(smem + OFF_WXR);
    float*  wxl   = (float*)(smem + OFF_WXL);
    float*  xlb   = (float*)(smem + OFF_XL);    // 2 x 32: (xs + noise_l)
    float*  xrb   = (float*)(smem + OFF_XR);    // 2 x 32: (xs + noise_r)

    const int tid    = threadIdx.x;
    const int lane   = tid & 31;
    const int wave   = tid >> 5;
    const int b_base = blockIdx.x * B_TILE;
    const float bro0 = b_ro[0];

    // ---- one-time LDS fill: W_full transposed -> bf16 ----------------------
    for (int idx = tid; idx < KP * KP; idx += 256) {
        const int k = idx / KP;          // GEMM K index (source hidden)
        const int j = idx % KP;          // GEMM N index (dest hidden)
        float v = 0.0f;
        if (k < H_DIM && j < H_DIM) {
            if (j < 2) v = (k < 2) ? w_ll[j * 2 + k] : w_rl[j * N_R + (k - 2)];
            else       v = (k < 2) ? w_lr[(j - 2) * 2 + k]
                                   : w_rr[(j - 2) * N_R + (k - 2)];
        }
        WT[k * KP + j] = (__bf16)v;
    }
    // zero both h buffers (padded columns must stay zero forever)
    for (int idx = tid; idx < 2 * HSLOT; idx += 256) {
        h_f[idx]  = 0.0f;
        h_bf[idx] = (__bf16)0.0f;
    }
    for (int j = tid; j < KP; j += 256) {
        bias[j] = (j < 2) ? b_ll[j] : ((j < H_DIM) ? b_rr[j - 2] : 0.0f);
        wro[j]  = (j < H_DIM) ? w_ro[j] : 0.0f;
    }
    for (int idx = tid; idx < N_R * 2; idx += 256) wxr[idx] = w_xr[idx];
    if (tid < 4) wxl[tid] = w_xl[tid];
    // stage noisy inputs for t = 0 into parity-0 xbuf
    if (wave == 0) {
        const int m = lane >> 1, c = lane & 1;
        const float x = xs[(((size_t)(b_base + m)) * T_TOT + 0) * 2 + c];
        const uint32_t base = (((uint32_t)(b_base + m)) << 1) | (uint32_t)c;
        xlb[lane] = x + SCL_NOISE * gauss(0x10000000u + base);
        xrb[lane] = x + SCL_NOISE * gauss(0x20000000u + base);
    }
    __syncthreads();

    // ---- tile assignment: wave w owns N-tiles {w, w+8, (w+16)} -------------
    int tlist[3];
    int tcount = 0;
    for (int nt = wave; nt < NT; nt += NWAVES) tlist[tcount++] = nt;

    const size_t ZOFF = (size_t)B_TOT * T_TOT * H_DIM;

    for (int t = 0; t < T_TOT; ++t) {
        const int p = t & 1;
        const __bf16* hb_cur = h_bf + p * HSLOT;
        __bf16*       hb_nxt = h_bf + (1 - p) * HSLOT;
        const float*  hf_cur = h_f + p * HSLOT;
        float*        hf_nxt = h_f + (1 - p) * HSLOT;
        const float*  xl_cur = xlb + p * 32;
        const float*  xr_cur = xrb + p * 32;
        float*        xl_nxt = xlb + (1 - p) * 32;
        float*        xr_nxt = xrb + (1 - p) * 32;

        // -- wave0: pre-stage noisy inputs for step t+1 ----------------------
        if (wave == 0 && t + 1 < T_TOT) {
            const int m = lane >> 1, c = lane & 1;
            const float x = xs[(((size_t)(b_base + m)) * T_TOT + (t + 1)) * 2 + c];
            const uint32_t base =
                (((uint32_t)(t + 1) * 512u + (uint32_t)(b_base + m)) << 1) | (uint32_t)c;
            xl_nxt[lane] = x + SCL_NOISE * gauss(0x10000000u + base);
            xr_nxt[lane] = x + SCL_NOISE * gauss(0x20000000u + base);
        }

        // -- per owned N-tile: WMMA GEMM + fused epilogue --------------------
        const int ma = lane & 15;
        const int kh = lane >> 4;
        const __bf16* hrow = hb_cur + ma * KP;
        for (int i = 0; i < tcount; ++i) {
            const int n0 = tlist[i] * 16;
            v8f acc = {};
#pragma unroll
            for (int k0 = 0; k0 < KP; k0 += 32) {
                // A fragment: 16-bit A 16x32 lane layout (ISA 7.12.2)
                v8bf alo = *(const v8bf*)(hrow + k0 + kh * 8);
                v8bf ahi = *(const v8bf*)(hrow + k0 + 16 + kh * 8);
                v16bf a;
#pragma unroll
                for (int e = 0; e < 8; ++e) { a[e] = alo[e]; a[8 + e] = ahi[e]; }
                // B fragment: col n0+(lane&15) per lane, K in same order as A
                const __bf16* wcol = WT + n0 + (lane & 15);
                v16bf b;
#pragma unroll
                for (int e = 0; e < 8; ++e) {
                    b[e]     = wcol[(k0 + kh * 8 + e) * KP];
                    b[8 + e] = wcol[(k0 + 16 + kh * 8 + e) * KP];
                }
                acc = __builtin_amdgcn_wmma_f32_16x16x32_bf16(
                        false, a, false, b, (short)0, acc, false, false);
            }
            // fused epilogue: pre -> tanh -> leaky update -> next h + hs out
            const int n = n0 + (lane & 15);
            if (n < H_DIM) {
                float wx0, wx1;
                if (n < 2) { wx0 = wxl[n * 2 + 0]; wx1 = wxl[n * 2 + 1]; }
                else       { wx0 = wxr[(n - 2) * 2 + 0]; wx1 = wxr[(n - 2) * 2 + 1]; }
                const float bj = bias[n];
                const float* xb = (n < 2) ? xl_cur : xr_cur;
                const int mo = (lane >> 4) * 8;
#pragma unroll
                for (int r = 0; r < 8; ++r) {
                    const int m = r + mo;
                    const float x_inj = xb[m * 2 + 0] * wx0 + xb[m * 2 + 1] * wx1;
                    const uint32_t zkey = 0x30000000u +
                        ((uint32_t)t * 512u + (uint32_t)(b_base + m)) * 258u + (uint32_t)n;
                    const float nz = SCL_NOISE * gauss(zkey);
                    const float pre = acc[r] + bj + x_inj + nz;
                    const float hold = hf_cur[m * KP + n];
                    const float hnew = (1.0f - A_LEAK) * hold + A_LEAK * fast_tanh(pre);
                    hf_nxt[m * KP + n] = hnew;
                    hb_nxt[m * KP + n] = (__bf16)hnew;
                    out[(((size_t)(b_base + m)) * T_TOT + t) * H_DIM + n] = hnew;
                }
            }
        }
        __syncthreads();   // fences h_new, next-step xbuf

        // -- readout zs: wave w reduces rows 2w, 2w+1 of the NEW state -------
#pragma unroll
        for (int rr = 0; rr < 2; ++rr) {
            const int m = 2 * wave + rr;
            float zp = 0.0f;
#pragma unroll
            for (int i = 0; i < KP / 32; ++i) {
                const int j = lane + 32 * i;
                zp += hf_nxt[m * KP + j] * wro[j];
            }
#pragma unroll
            for (int off = 16; off > 0; off >>= 1)
                zp += __shfl_xor(zp, off, 32);
            if (lane == 0)
                out[ZOFF + (size_t)(b_base + m) * T_TOT + t] = zp + bro0;
        }
        // safe without extra barrier: step t+1 writes only opposite-parity
        // buffers, and the barrier above fences everything it reads
    }
}

extern "C" void kernel_launch(void* const* d_in, const int* in_sizes, int n_in,
                              void* d_out, int out_size, void* d_ws, size_t ws_size,
                              hipStream_t stream) {
    (void)in_sizes; (void)n_in; (void)out_size; (void)d_ws; (void)ws_size;
    const float* xs   = (const float*)d_in[0];
    const float* w_ll = (const float*)d_in[1];
    const float* b_ll = (const float*)d_in[2];
    const float* w_rr = (const float*)d_in[3];
    const float* b_rr = (const float*)d_in[4];
    const float* w_lr = (const float*)d_in[5];
    const float* w_rl = (const float*)d_in[6];
    const float* w_xl = (const float*)d_in[7];
    const float* w_xr = (const float*)d_in[8];
    const float* w_ro = (const float*)d_in[9];
    const float* b_ro = (const float*)d_in[10];
    float* out = (float*)d_out;

    (void)hipFuncSetAttribute((const void*)two_hemi_rnn_kernel,
                              hipFuncAttributeMaxDynamicSharedMemorySize,
                              SMEM_BYTES);

    dim3 grid(B_TOT / B_TILE);   // 32 workgroups, one per 16-row batch tile
    dim3 block(256);             // 8 wave32s
    two_hemi_rnn_kernel<<<grid, block, SMEM_BYTES, stream>>>(
        xs, w_ll, b_ll, w_rr, b_rr, w_lr, w_rl, w_xl, w_xr, w_ro, b_ro, out);
}
